// Conv2dShift_46866683134640
// MI455X (gfx1250) — matrix operand order repaired and checked by reading, not compiled, not run on hardware
//
#include <hip/hip_runtime.h>
#include <hip/hip_bf16.h>

typedef __attribute__((ext_vector_type(16))) _Float16 v16h;
typedef __attribute__((ext_vector_type(8)))  float    v8f;

#define CIN    256
#define COUT   256
#define HH     56
#define WW     56
#define HWSZ   3136        // 56*56
#define KTOT   2304        // 256*9
#define NPIX   100352      // 32*3136
#define IMGSZ  802816      // 256*3136
#define STAGES 72          // KTOT / 32
#define TILEU  2048        // u32 per A stage tile (128 M x 32 K halves)

static __device__ __forceinline__ unsigned short h2u(_Float16 x) {
    union { _Float16 f; unsigned short u; } c; c.f = x; return c.u;
}

// ---------------------------------------------------------------------------
// Pass 1: power-of-two weights in f16, pre-arranged in WMMA fragment-major
// order per (M-tile, K-stage): each stage tile is a contiguous 8KB block that
// the conv kernel can stream straight into LDS with async copies.
//   u32 index: (mt*STAGES + s)*TILEU + (h*128 + m)*8 + v
//   K of packed pair: s*32 + (v<4?0:16) + 2*(v&3) + 8h + {0,1}
// ---------------------------------------------------------------------------
__global__ __launch_bounds__(256) void pack_weights(const float* __restrict__ shift,
                                                    const float* __restrict__ sgn,
                                                    unsigned int* __restrict__ w2u) {
    int u = blockIdx.x * 256 + threadIdx.x;             // 0 .. 294911
    if (u >= 2 * STAGES * TILEU) return;
    int mt = u / (STAGES * TILEU);
    int r1 = u - mt * (STAGES * TILEU);
    int s  = r1 >> 11;
    int r2 = r1 & 2047;
    int h  = r2 >> 10;
    int r3 = r2 & 1023;
    int m  = r3 >> 3;
    int v  = r3 & 7;
    int kbase = s * 32 + ((v & 4) ? 16 : 0) + 2 * (v & 3) + 8 * h;
    int co    = mt * 128 + m;
    unsigned int pr = 0;
    #pragma unroll
    for (int j = 0; j < 2; ++j) {
        int in  = co * KTOT + (kbase + j);              // [co][ci][kh][kw] flat
        float sv = shift[in];
        sv = fminf(fmaxf(sv, -14.0f), 0.0f);
        float sr = rintf(sv);                           // round half-to-even
        float g  = rintf(sgn[in]);
        float sg = (g > 0.0f) ? 1.0f : ((g < 0.0f) ? -1.0f : 0.0f);
        float vv = exp2f(sr) * sg;                      // exact in f16 (s >= -14)
        pr |= ((unsigned)h2u((_Float16)vv)) << (16 * j);
    }
    w2u[u] = pr;
}

// ---------------------------------------------------------------------------
// Pass 2: implicit-GEMM conv via V_WMMA_F32_16X16X32_F16, split-f16 activations
//   grid: (1568 N-tiles, 2 M-tiles), block: 256 threads (8 wave32)
//   block tile: M=128 x N=64, K chunk = 32, LDS ping-pong
//   A tile: global_load_async_to_lds_b128 (ASYNCcnt), B tile: VALU split path
// ---------------------------------------------------------------------------
__global__ __launch_bounds__(256) void conv_shift_wmma(const float* __restrict__ X,
                                                       const unsigned int* __restrict__ Wf,
                                                       const float* __restrict__ bias,
                                                       float* __restrict__ out) {
    // fragment-major LDS: each lane's whole v16h fragment is 32B contiguous
    __shared__ unsigned int As[2][TILEU];  // weight tile (async-copied)
    __shared__ unsigned int Bh[2][1024];   // [(h*64+n)*8 + v]  activation hi
    __shared__ unsigned int Bl[2][1024];   //                    activation lo
    __shared__ float biasq[128];

    const int t     = threadIdx.x;
    const int ntile = blockIdx.x * 64;
    const int mt    = blockIdx.y;
    const int mtile = mt * 128;

    if (t < 128) {
        float bv = bias[mtile + t];
        bv = fminf(fmaxf(floorf(bv * 65536.0f) * (1.0f / 65536.0f),
                         -32768.0f), 32767.0f);
        biasq[t] = bv;
    }

    // ---- per-thread fill geometry (stage-invariant) ----
    const int kp     = t >> 4;            // k-pair slot 0..15
    const int klocal = kp << 1;           // even k within 32-chunk
    const int nb     = (t & 15) << 2;     // 4 consecutive pixels
    // B-fragment layout (16-bit B 32x16): K = (v<4?0:16) + 2*(v&3) + 8h + j
    const int hB = (klocal >> 3) & 1;
    const int vB = ((klocal >> 4) << 2) | ((klocal >> 1) & 3);

    int pbase[4], poh[4], pow_[4];
    #pragma unroll
    for (int n = 0; n < 4; ++n) {
        int p  = ntile + nb + n;
        int b  = p / HWSZ;
        int hw = p - b * HWSZ;
        int oh = hw / WW;
        int ow = hw - oh * WW;
        poh[n]   = oh;
        pow_[n]  = ow;
        pbase[n] = b * IMGSZ + oh * WW + ow;
    }

    // ---- compute geometry ----
    const int lane = t & 31;
    const int wid  = t >> 5;
    const int wm   = (wid & 3) * 32;      // wave M offset (4 waves over 128)
    const int wn   = (wid >> 2) * 32;     // wave N offset (2 waves over 64)
    const int lh   = lane >> 4;
    const int ln   = lane & 15;
    const int offA0 = (lh * 128 + wm + ln) * 8;
    const int offA1 = offA0 + 16 * 8;
    const int offB0 = (lh * 64 + wn + ln) * 8;
    const int offB1 = offB0 + 16 * 8;

    v8f acc[2][2];
    #pragma unroll
    for (int i = 0; i < 2; ++i)
        #pragma unroll
        for (int j = 0; j < 2; ++j)
            acc[i][j] = (v8f){0.f, 0.f, 0.f, 0.f, 0.f, 0.f, 0.f, 0.f};

    const unsigned int* Wbase = Wf + mt * (STAGES * TILEU);

    // ---- async A-tile fill: memory -> LDS, no VGPR round trip (ASYNCcnt) ----
    auto asyncA = [&](int s1, int bufi) {
        unsigned lds0 = (unsigned)(uintptr_t)&As[bufi][t * 8];
        const unsigned int* g0 = Wbase + s1 * TILEU + t * 8;
        // inst offset applies to both LDS and global side (ISA 10.x pseudocode)
        asm volatile("global_load_async_to_lds_b128 %0, %1, off"
                     :: "v"(lds0), "v"(g0) : "memory");
        asm volatile("global_load_async_to_lds_b128 %0, %1, off offset:16"
                     :: "v"(lds0), "v"(g0) : "memory");
    };

    // ---- global gather of activations for one K-chunk ----
    auto loadg = [&](int k0, float* xr) {
        const int kk = k0 + klocal;
        #pragma unroll
        for (int e = 0; e < 2; ++e) {
            int kg = kk + e;
            int ci = kg / 9;
            int rr = kg - ci * 9;
            int kh = rr / 3;
            int kw = rr - kh * 3;
            int coff = ci * HWSZ + (kh - 1) * WW + (kw - 1);
            #pragma unroll
            for (int n = 0; n < 4; ++n) {
                int ih = poh[n] + kh - 1;
                int iw = pow_[n] + kw - 1;
                float xv = 0.0f;
                if ((unsigned)ih < (unsigned)HH && (unsigned)iw < (unsigned)WW)
                    xv = X[pbase[n] + coff];
                xr[e * 4 + n] = xv;
            }
        }
    };

    // ---- fixed-point quantize + f16 hi/lo split + LDS stores ----
    auto stolds = [&](int bufi, const float* xr) {
        unsigned short hs[8], ls[8];
        #pragma unroll
        for (int q = 0; q < 8; ++q) {
            float f = floorf(xr[q] * 65536.0f) * (1.0f / 65536.0f);
            f = fminf(fmaxf(f, -32768.0f), 32767.0f);
            _Float16 hi = (_Float16)f;
            _Float16 lo = (_Float16)(f - (float)hi);
            hs[q] = h2u(hi);
            ls[q] = h2u(lo);
        }
        #pragma unroll
        for (int n = 0; n < 4; ++n) {
            int off = (hB * 64 + nb + n) * 8 + vB;
            Bh[bufi][off] = (unsigned)hs[n] | ((unsigned)hs[4 + n] << 16);
            Bl[bufi][off] = (unsigned)ls[n] | ((unsigned)ls[4 + n] << 16);
        }
    };

    auto compute = [&](int bufi) {
        const v16h a0  = *(const v16h*)&As[bufi][offA0];
        const v16h a1  = *(const v16h*)&As[bufi][offA1];
        const v16h bh0 = *(const v16h*)&Bh[bufi][offB0];
        const v16h bh1 = *(const v16h*)&Bh[bufi][offB1];
        const v16h bl0 = *(const v16h*)&Bl[bufi][offB0];
        const v16h bl1 = *(const v16h*)&Bl[bufi][offB1];
        acc[0][0] = __builtin_amdgcn_wmma_f32_16x16x32_f16(false, a0, false, bh0, (short)0, acc[0][0], false, false);
        acc[0][0] = __builtin_amdgcn_wmma_f32_16x16x32_f16(false, a0, false, bl0, (short)0, acc[0][0], false, false);
        acc[0][1] = __builtin_amdgcn_wmma_f32_16x16x32_f16(false, a0, false, bh1, (short)0, acc[0][1], false, false);
        acc[0][1] = __builtin_amdgcn_wmma_f32_16x16x32_f16(false, a0, false, bl1, (short)0, acc[0][1], false, false);
        acc[1][0] = __builtin_amdgcn_wmma_f32_16x16x32_f16(false, a1, false, bh0, (short)0, acc[1][0], false, false);
        acc[1][0] = __builtin_amdgcn_wmma_f32_16x16x32_f16(false, a1, false, bl0, (short)0, acc[1][0], false, false);
        acc[1][1] = __builtin_amdgcn_wmma_f32_16x16x32_f16(false, a1, false, bh1, (short)0, acc[1][1], false, false);
        acc[1][1] = __builtin_amdgcn_wmma_f32_16x16x32_f16(false, a1, false, bl1, (short)0, acc[1][1], false, false);
    };

    // ---- prologue: stage 0 into buffer 0 ----
    {
        asyncA(0, 0);
        float xr[8];
        loadg(0, xr);
        stolds(0, xr);
        asm volatile("s_wait_asynccnt 0" ::: "memory");
    }
    __syncthreads();

    // ---- main K loop: 72 stages of 32, ping-pong double buffer ----
    for (int s = 0; s < STAGES; ++s) {
        const int cur  = s & 1;
        const int nxt  = cur ^ 1;
        const bool more = (s + 1) < STAGES;
        float xr[8] = {};
        if (more) {
            asyncA(s + 1, nxt);                          // LDS-direct, ASYNCcnt
            loadg((s + 1) * 32, xr);                     // activation gather
        }
        compute(cur);                                    // 8x v_wmma_f32_16x16x32_f16
        if (more)
            stolds(nxt, xr);
        asm volatile("s_wait_asynccnt 0" ::: "memory");  // A tile landed in LDS
        __syncthreads();
    }

    // ---- epilogue: C layout VGPR r -> M = r + 8h, lane -> N ----
    #pragma unroll
    for (int i = 0; i < 2; ++i) {
        #pragma unroll
        for (int j = 0; j < 2; ++j) {
            int p  = ntile + wn + j * 16 + ln;
            int b  = p / HWSZ;
            int hw = p - b * HWSZ;
            int obase = b * IMGSZ + hw;
            #pragma unroll
            for (int r = 0; r < 8; ++r) {
                int col = wm + i * 16 + r + 8 * lh;      // local cout
                float vv = acc[i][j][r] + biasq[col];
                out[obase + (mtile + col) * HWSZ] = vv;
            }
        }
    }
}

// ---------------------------------------------------------------------------
extern "C" void kernel_launch(void* const* d_in, const int* in_sizes, int n_in,
                              void* d_out, int out_size, void* d_ws, size_t ws_size,
                              hipStream_t stream) {
    const float* X     = (const float*)d_in[0];   // (32,256,56,56)
    const float* shift = (const float*)d_in[1];   // (256,256,3,3)
    const float* sgn   = (const float*)d_in[2];   // (256,256,3,3)
    const float* bias  = (const float*)d_in[3];   // (256,)
    float*       out   = (float*)d_out;           // (32,256,56,56)

    unsigned int* W2 = (unsigned int*)d_ws;       // 294912 u32 = 1.125 MB

    pack_weights<<<(2 * STAGES * TILEU + 255) / 256, 256, 0, stream>>>(shift, sgn, W2);

    dim3 grid(NPIX / 64, COUT / 128);             // (1568, 2)
    conv_shift_wmma<<<grid, 256, 0, stream>>>(X, W2, bias, out);
}
